// TtMoeLayer_7172595384493
// MI455X (gfx1250) — matrix operand order repaired and checked by reading, not compiled, not run on hardware
//
#include <hip/hip_runtime.h>

// MI455X (gfx1250) dense-MoE SwiGLU.
// wave32, v_wmma_f32_16x16x32_bf16 (f32 accum), double-buffered async
// global->LDS staging (ASYNCcnt), pre-transposed bf16 weight operands.

typedef __bf16 bf16;
typedef __attribute__((ext_vector_type(8)))  __bf16 v8bf;
typedef __attribute__((ext_vector_type(16))) __bf16 v16bf;
typedef __attribute__((ext_vector_type(8)))  float  v8f;

union Frag16 { v16bf v; v8bf h[2]; };

#define T_DIM 8192
#define H_DIM 4096
#define F_DIM 2048
#define E_DIM 8

#define BM  128
#define BN  64
#define BK  32
#define LDK 40    // padded LDS stride: 80B, 16B-aligned, bank-conflict-free

__device__ __forceinline__ unsigned short f2bfu(float f) {
  unsigned u = __float_as_uint(f);
  u += 0x7FFFu + ((u >> 16) & 1u);           // round-to-nearest-even
  return (unsigned short)(u >> 16);
}
__device__ __forceinline__ bf16 f2bf(float f) {
  unsigned short s = f2bfu(f);
  return __builtin_bit_cast(bf16, s);
}

// LDS byte offset of a shared-memory pointer (LDS aperture keeps offset in addr[31:0])
__device__ __forceinline__ unsigned ldsoff(const void* p) {
  return (unsigned)(unsigned long long)p;
}
// CDNA5 async copy: 16B per lane, global -> LDS, tracked by ASYNCcnt
__device__ __forceinline__ void async_copy_b128(unsigned lds_addr, const void* gaddr) {
  asm volatile("global_load_async_to_lds_b128 %0, %1, off"
               :: "v"(lds_addr), "v"(gaddr)
               : "memory");
}
__device__ __forceinline__ void wait_async0() {
  asm volatile("s_wait_asynccnt 0" ::: "memory");
}

// ---------------------------------------------------------------------------
// Gating: logits = x @ gate_w, top-2, softmax -> dense combine[T,E]
// ---------------------------------------------------------------------------
__global__ __launch_bounds__(256) void gate_topk_kernel(
    const float* __restrict__ x, const float* __restrict__ gw,
    float* __restrict__ combine)
{
  const int lane = threadIdx.x & 31;
  const int wave = threadIdx.x >> 5;
  const int t = blockIdx.x * 8 + wave;
  const float* xr = x + (size_t)t * H_DIM;

  float acc[E_DIM];
#pragma unroll
  for (int e = 0; e < E_DIM; ++e) acc[e] = 0.f;

  for (int h = lane; h < H_DIM; h += 32) {
    float xv = xr[h];
    const float* g = gw + (size_t)h * E_DIM;
#pragma unroll
    for (int e = 0; e < E_DIM; ++e) acc[e] += xv * g[e];
  }
#pragma unroll
  for (int e = 0; e < E_DIM; ++e)
    for (int off = 16; off > 0; off >>= 1)
      acc[e] += __shfl_xor(acc[e], off, 32);

  if (lane == 0) {
    int i0 = 0; float v0 = acc[0];
#pragma unroll
    for (int e = 1; e < E_DIM; ++e) if (acc[e] > v0) { v0 = acc[e]; i0 = e; }
    int i1 = -1; float v1 = -3.0e38f;
#pragma unroll
    for (int e = 0; e < E_DIM; ++e) if (e != i0 && acc[e] > v1) { v1 = acc[e]; i1 = e; }
    float p0 = 1.f / (1.f + __expf(v1 - v0));
    float p1 = 1.f - p0;
#pragma unroll
    for (int e = 0; e < E_DIM; ++e)
      combine[(size_t)t * E_DIM + e] = (e == i0) ? p0 : ((e == i1) ? p1 : 0.f);
  }
}

// ---------------------------------------------------------------------------
// x (fp32) -> xb (bf16)
// ---------------------------------------------------------------------------
__global__ __launch_bounds__(256) void cvt_bf16_kernel(
    const float* __restrict__ x, bf16* __restrict__ xb)
{
  size_t i = ((size_t)blockIdx.x * 256 + threadIdx.x) * 4;
  float4 f = *(const float4*)(x + i);
  unsigned p0 = (unsigned)f2bfu(f.x) | ((unsigned)f2bfu(f.y) << 16);
  unsigned p1 = (unsigned)f2bfu(f.z) | ((unsigned)f2bfu(f.w) << 16);
  *(uint2*)(xb + i) = make_uint2(p0, p1);
}

// ---------------------------------------------------------------------------
// Weight prep (once per expert, streaming): fp32 [R,C] -> bf16 [C,R].
// ---------------------------------------------------------------------------
__global__ __launch_bounds__(256) void transpose_cvt_kernel(
    const float* __restrict__ src,   // [R, C]
    bf16*        __restrict__ dst,   // [C, R]
    int R, int C)
{
  __shared__ bf16 tile[32][34];
  const int tx = threadIdx.x & 31;
  const int ty = threadIdx.x >> 5;
  const int c0 = blockIdx.x * 32;
  const int r0 = blockIdx.y * 32;
#pragma unroll
  for (int i = 0; i < 32; i += 8)
    tile[ty + i][tx] = f2bf(src[(size_t)(r0 + ty + i) * C + c0 + tx]);
  __syncthreads();
#pragma unroll
  for (int i = 0; i < 32; i += 8)
    dst[(size_t)(c0 + ty + i) * R + r0 + tx] = tile[tx][ty + i];
}

// ---------------------------------------------------------------------------
// h = silu(x@w1e) * (x@w3e) -> hb (bf16)  [T,F]
// Block 128x64, 8 waves = 4(M)x2(N), wave tile 32x32 (2x2 WMMA, x2 matrices).
// Double-buffered async staging: stage tile k+1 while computing tile k.
// ---------------------------------------------------------------------------
__global__ __launch_bounds__(256) void ffn_gate_up_kernel(
    const bf16* __restrict__ xb,    // [T,H] bf16
    const bf16* __restrict__ w1t,   // [F,H] bf16 (N-major)
    const bf16* __restrict__ w3t,   // [F,H] bf16
    bf16*       __restrict__ hb)    // [T,F] bf16
{
  __shared__ __align__(16) bf16 As [2][BM][LDK];
  __shared__ __align__(16) bf16 B1s[2][BN][LDK];
  __shared__ __align__(16) bf16 B3s[2][BN][LDK];

  const int tid  = threadIdx.x;
  const int lane = tid & 31;
  const int wave = tid >> 5;
  const int wm   = wave >> 1;       // 0..3 : 32-row group
  const int wn   = wave & 1;        // 0..1 : 32-col group
  const int m0   = blockIdx.x * BM;
  const int n0   = blockIdx.y * BN;

  v8f acc1[2][2], acc3[2][2];       // [ms][ns]
#pragma unroll
  for (int ms = 0; ms < 2; ++ms)
#pragma unroll
    for (int ns = 0; ns < 2; ++ns)
#pragma unroll
      for (int i = 0; i < 8; ++i) { acc1[ms][ns][i] = 0.f; acc3[ms][ns][i] = 0.f; }

  // staging: 256 threads x 8 bf16 (16B) = 64 rows x 32 cols per pass
  const int sr = (tid * 8) >> 5;    // 0..63
  const int sc = (tid * 8) & 31;    // 0,8,16,24

  auto stage = [&](int b, int k0) {
    async_copy_b128(ldsoff(&As [b][sr     ][sc]), xb  + (size_t)(m0 + sr     ) * H_DIM + k0 + sc);
    async_copy_b128(ldsoff(&As [b][sr + 64][sc]), xb  + (size_t)(m0 + sr + 64) * H_DIM + k0 + sc);
    async_copy_b128(ldsoff(&B1s[b][sr     ][sc]), w1t + (size_t)(n0 + sr     ) * H_DIM + k0 + sc);
    async_copy_b128(ldsoff(&B3s[b][sr     ][sc]), w3t + (size_t)(n0 + sr     ) * H_DIM + k0 + sc);
  };

  stage(0, 0);
  int buf = 0;
  for (int k0 = 0; k0 < H_DIM; k0 += BK, buf ^= 1) {
    wait_async0();              // my stage for this tile is done
    __syncthreads();            // everyone's stage done; prev compute done
    if (k0 + BK < H_DIM) stage(buf ^ 1, k0 + BK);   // overlap with compute

    Frag16 fa[2], fb1[2], fb3[2];
#pragma unroll
    for (int ms = 0; ms < 2; ++ms) {
      const int row = wm * 32 + ms * 16 + (lane & 15);
      const int cb  = (lane >> 4) << 4;             // lanes 0-15: K0-15; 16-31: K16-31
      fa[ms].h[0] = *(const v8bf*)&As[buf][row][cb];
      fa[ms].h[1] = *(const v8bf*)&As[buf][row][cb + 8];
    }
#pragma unroll
    for (int ns = 0; ns < 2; ++ns) {
      const int nc = wn * 32 + ns * 16 + (lane & 15);
      const int kb = (lane >> 4) << 4;              // lanes 0-15: K0-15; 16-31: K16-31
      fb1[ns].h[0] = *(const v8bf*)&B1s[buf][nc][kb];
      fb1[ns].h[1] = *(const v8bf*)&B1s[buf][nc][kb + 8];
      fb3[ns].h[0] = *(const v8bf*)&B3s[buf][nc][kb];
      fb3[ns].h[1] = *(const v8bf*)&B3s[buf][nc][kb + 8];
    }
#pragma unroll
    for (int ms = 0; ms < 2; ++ms)
#pragma unroll
      for (int ns = 0; ns < 2; ++ns) {
        acc1[ms][ns] = __builtin_amdgcn_wmma_f32_16x16x32_bf16(
            false, fa[ms].v, false, fb1[ns].v, (short)0, acc1[ms][ns], false, false);
        acc3[ms][ns] = __builtin_amdgcn_wmma_f32_16x16x32_bf16(
            false, fa[ms].v, false, fb3[ns].v, (short)0, acc3[ms][ns], false, false);
      }
  }

  // epilogue: h = a*sigmoid(a)*b with fast v_rcp_f32
#pragma unroll
  for (int ms = 0; ms < 2; ++ms) {
    const int rbase = m0 + wm * 32 + ms * 16 + ((lane >> 4) << 3);
#pragma unroll
    for (int ns = 0; ns < 2; ++ns) {
      const int col = n0 + wn * 32 + ns * 16 + (lane & 15);
#pragma unroll
      for (int v = 0; v < 8; ++v) {
        float a = acc1[ms][ns][v];
        float b = acc3[ms][ns][v];
        float hval = a * __builtin_amdgcn_rcpf(1.f + __expf(-a)) * b;
        hb[(size_t)(rbase + v) * F_DIM + col] = f2bf(hval);
      }
    }
  }
}

// ---------------------------------------------------------------------------
// out (+)= combine[:,e] * (hb @ w2e)   [T,H]
// ---------------------------------------------------------------------------
__global__ __launch_bounds__(256) void moe_down_kernel(
    const bf16*  __restrict__ hb,    // [T,F] bf16
    const bf16*  __restrict__ w2t,   // [H,F] bf16 (N-major)
    const float* __restrict__ comb,  // [T,E]
    float*       __restrict__ out,   // [T,H]
    int expert, int accumulate)
{
  __shared__ __align__(16) bf16 As[2][BM][LDK];
  __shared__ __align__(16) bf16 Bs[2][BN][LDK];

  const int tid  = threadIdx.x;
  const int lane = tid & 31;
  const int wave = tid >> 5;
  const int wm   = wave >> 1;
  const int wn   = wave & 1;
  const int m0   = blockIdx.x * BM;
  const int n0   = blockIdx.y * BN;

  v8f acc[2][2];
#pragma unroll
  for (int ms = 0; ms < 2; ++ms)
#pragma unroll
    for (int ns = 0; ns < 2; ++ns)
#pragma unroll
      for (int i = 0; i < 8; ++i) acc[ms][ns][i] = 0.f;

  const int sr = (tid * 8) >> 5;
  const int sc = (tid * 8) & 31;

  auto stage = [&](int b, int k0) {
    async_copy_b128(ldsoff(&As[b][sr     ][sc]), hb  + (size_t)(m0 + sr     ) * F_DIM + k0 + sc);
    async_copy_b128(ldsoff(&As[b][sr + 64][sc]), hb  + (size_t)(m0 + sr + 64) * F_DIM + k0 + sc);
    async_copy_b128(ldsoff(&Bs[b][sr     ][sc]), w2t + (size_t)(n0 + sr     ) * F_DIM + k0 + sc);
  };

  stage(0, 0);
  int buf = 0;
  for (int k0 = 0; k0 < F_DIM; k0 += BK, buf ^= 1) {
    wait_async0();
    __syncthreads();
    if (k0 + BK < F_DIM) stage(buf ^ 1, k0 + BK);

    Frag16 fa[2], fb[2];
#pragma unroll
    for (int ms = 0; ms < 2; ++ms) {
      const int row = wm * 32 + ms * 16 + (lane & 15);
      const int cb  = (lane >> 4) << 4;
      fa[ms].h[0] = *(const v8bf*)&As[buf][row][cb];
      fa[ms].h[1] = *(const v8bf*)&As[buf][row][cb + 8];
    }
#pragma unroll
    for (int ns = 0; ns < 2; ++ns) {
      const int nc = wn * 32 + ns * 16 + (lane & 15);
      const int kb = (lane >> 4) << 4;
      fb[ns].h[0] = *(const v8bf*)&Bs[buf][nc][kb];
      fb[ns].h[1] = *(const v8bf*)&Bs[buf][nc][kb + 8];
    }
#pragma unroll
    for (int ms = 0; ms < 2; ++ms)
#pragma unroll
      for (int ns = 0; ns < 2; ++ns)
        acc[ms][ns] = __builtin_amdgcn_wmma_f32_16x16x32_bf16(
            false, fa[ms].v, false, fb[ns].v, (short)0, acc[ms][ns], false, false);
  }

#pragma unroll
  for (int ms = 0; ms < 2; ++ms) {
    const int rbase = m0 + wm * 32 + ms * 16 + ((lane >> 4) << 3);
#pragma unroll
    for (int v = 0; v < 8; ++v) {
      const int row = rbase + v;
      const float cv = comb[(size_t)row * E_DIM + expert];
#pragma unroll
      for (int ns = 0; ns < 2; ++ns) {
        const int col = n0 + wn * 32 + ns * 16 + (lane & 15);
        const size_t o = (size_t)row * H_DIM + col;
        const float r = cv * acc[ms][ns][v];
        out[o] = accumulate ? (out[o] + r) : r;
      }
    }
  }
}

// ---------------------------------------------------------------------------
extern "C" void kernel_launch(void* const* d_in, const int* in_sizes, int n_in,
                              void* d_out, int out_size, void* d_ws, size_t ws_size,
                              hipStream_t stream)
{
  (void)in_sizes; (void)n_in; (void)out_size; (void)ws_size;
  const float* x  = (const float*)d_in[0];   // [T,H]
  const float* gw = (const float*)d_in[1];   // [H,E]
  const float* w1 = (const float*)d_in[2];   // [E,H,F]
  const float* w3 = (const float*)d_in[3];   // [E,H,F]
  const float* w2 = (const float*)d_in[4];   // [E,F,H]
  float* out = (float*)d_out;                // [T,H]

  char* ws = (char*)d_ws;
  size_t off = 0;
  float* combine = (float*)ws;                               // 256 KB
  off += (((size_t)T_DIM * E_DIM * sizeof(float)) + 255) & ~(size_t)255;
  bf16* xb  = (bf16*)(ws + off); off += (size_t)T_DIM * H_DIM * sizeof(bf16);  // 64 MB
  bf16* hbf = (bf16*)(ws + off); off += (size_t)T_DIM * F_DIM * sizeof(bf16);  // 32 MB
  bf16* w1t = (bf16*)(ws + off); off += (size_t)F_DIM * H_DIM * sizeof(bf16);  // 16.8 MB
  bf16* w3t = (bf16*)(ws + off); off += (size_t)F_DIM * H_DIM * sizeof(bf16);
  bf16* w2t = (bf16*)(ws + off); off += (size_t)H_DIM * F_DIM * sizeof(bf16);

  gate_topk_kernel<<<T_DIM / 8, 256, 0, stream>>>(x, gw, combine);
  cvt_bf16_kernel<<<(unsigned)(((size_t)T_DIM * H_DIM) / 1024), 256, 0, stream>>>(x, xb);

  for (int e = 0; e < E_DIM; ++e) {
    const float* w1e = w1 + (size_t)e * H_DIM * F_DIM;   // [H,F]
    const float* w3e = w3 + (size_t)e * H_DIM * F_DIM;   // [H,F]
    const float* w2e = w2 + (size_t)e * F_DIM * H_DIM;   // [F,H]
    transpose_cvt_kernel<<<dim3(F_DIM / 32, H_DIM / 32), 256, 0, stream>>>(w1e, w1t, H_DIM, F_DIM);
    transpose_cvt_kernel<<<dim3(F_DIM / 32, H_DIM / 32), 256, 0, stream>>>(w3e, w3t, H_DIM, F_DIM);
    transpose_cvt_kernel<<<dim3(H_DIM / 32, F_DIM / 32), 256, 0, stream>>>(w2e, w2t, F_DIM, H_DIM);

    ffn_gate_up_kernel<<<dim3(T_DIM / BM, F_DIM / BN), 256, 0, stream>>>(xb, w1t, w3t, hbf);
    moe_down_kernel<<<dim3(T_DIM / BM, H_DIM / BN), 256, 0, stream>>>(hbf, w2t, combine, out, e, e > 0);
  }
}
